// CNN_85976655331672
// MI455X (gfx1250) — 8-bit matrix operand order repaired and checked by reading, compiled, not run on hardware
//
#include <hip/hip_runtime.h>


typedef int v8i __attribute__((ext_vector_type(8)));
typedef int v4i_g __attribute__((vector_size(16)));

#define HDIM 56
#define WDIM 56
#define NIMG 32
#define CIN 64
#define COUT 128
#define HW (HDIM * WDIM)
#define MTOT (NIMG * HW)
#define KTILES 9
#define BTILE (COUT * CIN)

#if defined(__has_builtin)
#  if __has_builtin(__builtin_amdgcn_global_load_async_to_lds_b128)
#    define HAVE_ASYNC_LDS 1
#  endif
#endif
#ifndef HAVE_ASYNC_LDS
#  define HAVE_ASYNC_LDS 0
#endif

typedef __attribute__((address_space(1))) v4i_g as1_v4i;
typedef __attribute__((address_space(3))) v4i_g as3_v4i;

__device__ __forceinline__ as1_v4i* as_g4(const void* p) {
  return (as1_v4i*)(unsigned long long)p;
}
__device__ __forceinline__ as3_v4i* as_l4(void* p) {
  return (as3_v4i*)(unsigned int)(unsigned long long)p;
}

__device__ __forceinline__ void stage_wait_async() {
#if HAVE_ASYNC_LDS
#  if __has_builtin(__builtin_amdgcn_s_wait_asynccnt)
  __builtin_amdgcn_s_wait_asynccnt(0);
#  else
  asm volatile("s_wait_asynccnt 0x0" ::: "memory");
#  endif
#endif
}

__device__ __forceinline__ void stage_b(const signed char* __restrict__ wT2,
                                        int ki, char* buf, int tid) {
#if HAVE_ASYNC_LDS
  const signed char* g = wT2 + (long)ki * BTILE + tid * 16;
  __builtin_amdgcn_global_load_async_to_lds_b128(as_g4(g),
                                                 as_l4(buf + tid * 16), 0, 0);
  __builtin_amdgcn_global_load_async_to_lds_b128(as_g4(g + 4096),
                                                 as_l4(buf + 4096 + tid * 16), 0, 0);
#else
  const int4* src = (const int4*)(wT2 + (long)ki * BTILE);
  int4* dst = (int4*)buf;
  dst[tid]       = src[tid];
  dst[tid + 256] = src[tid + 256];
#endif
}

__device__ __forceinline__ signed char quant8(float v, float inv_step) {
  float r = __builtin_rintf(v * inv_step);
  r = fminf(fmaxf(r, -127.0f), 127.0f);
  return (signed char)(int)r;
}

__global__ void __launch_bounds__(256) quant_x_kernel(
    const float* __restrict__ x, const float* __restrict__ alpha,
    signed char* __restrict__ ixT) {
  int i = blockIdx.x * 256 + threadIdx.x;
  if (i >= NIMG * HW * CIN) return;
  float inv_step = 127.0f / fmaxf(alpha[0], 0.0f);
  int c  = i & 63;
  int p  = i >> 6;
  int n  = p / HW;
  int yx = p - n * HW;
  ixT[i] = quant8(x[((long)n * CIN + c) * HW + yx], inv_step);
}

__global__ void __launch_bounds__(256) quant_w_kernel(
    const float* __restrict__ w, const float* __restrict__ alpha,
    signed char* __restrict__ wT2) {
  int i = blockIdx.x * 256 + threadIdx.x;
  if (i >= KTILES * COUT * CIN) return;
  float inv_step = 127.0f / fmaxf(alpha[0], 0.0f);
  int c  = i & 63;
  int o  = (i >> 6) & 127;
  int ki = i >> 13;
  int r = ki / 3, s = ki - r * 3;
  wT2[i] = quant8(w[(((long)o * CIN + c) * 3 + r) * 3 + s], inv_step);
}

__device__ __forceinline__ v8i load_a_frag(const signed char* __restrict__ pc,
                                           int y, int x, int dr, int ds,
                                           int half) {
  bool inb = ((unsigned)(y + dr) < (unsigned)HDIM) &&
             ((unsigned)(x + ds) < (unsigned)WDIM);
  const signed char* p = pc + (dr * WDIM + ds) * 64 + half * 16;
  unsigned long long d0 = 0, d1 = 0, d2 = 0, d3 = 0;
  if (inb) {
    d0 = *(const unsigned long long*)(p + 0);
    d1 = *(const unsigned long long*)(p + 8);
    d2 = *(const unsigned long long*)(p + 32);
    d3 = *(const unsigned long long*)(p + 40);
  }
  v8i a;
  a[0] = (int)d0; a[1] = (int)(d0 >> 32);
  a[2] = (int)d1; a[3] = (int)(d1 >> 32);
  a[4] = (int)d2; a[5] = (int)(d2 >> 32);
  a[6] = (int)d3; a[7] = (int)(d3 >> 32);
  return a;
}

__device__ __forceinline__ v8i load_b_frag(const char* ldsB, int col, int half) {
  int4 lo = *(const int4*)(ldsB + col * 64 + half * 16);
  int4 hi = *(const int4*)(ldsB + col * 64 + 32 + half * 16);
  v8i b;
  b[0] = lo.x; b[1] = lo.y; b[2] = lo.z; b[3] = lo.w;
  b[4] = hi.x; b[5] = hi.y; b[6] = hi.z; b[7] = hi.w;
  return b;
}

__global__ void __launch_bounds__(256) conv_wmma_kernel(
    const signed char* __restrict__ ixT, const signed char* __restrict__ wT2,
    const float* __restrict__ alpha_x, const float* __restrict__ alpha_w,
    float* __restrict__ out) {
  __shared__ char ldsB[2][BTILE];

  const int tid   = threadIdx.x;
  const int lane  = tid & 31;
  const int wave  = tid >> 5;
  const int mWave = wave & 3;
  const int nWave = wave >> 2;
  const int row   = lane & 15;
  const int half  = lane >> 4;
  const int block_m0 = blockIdx.x * 128;

  int m0 = block_m0 + mWave * 32 + row;
  int m1 = m0 + 16;
  int n0 = m0 / HW; int t0 = m0 - n0 * HW; int y0 = t0 / WDIM; int x0 = t0 - y0 * WDIM;
  int n1 = m1 / HW; int t1 = m1 - n1 * HW; int y1 = t1 / WDIM; int x1 = t1 - y1 * WDIM;
  const signed char* pc0 = ixT + ((long)n0 * HW + (long)y0 * WDIM + x0) * 64;
  const signed char* pc1 = ixT + ((long)n1 * HW + (long)y1 * WDIM + x1) * 64;

  v8i acc[2][4] = {};

  stage_b(wT2, 0, ldsB[0], tid);
  stage_wait_async();
  __syncthreads();

#pragma unroll
  for (int ki = 0; ki < KTILES; ++ki) {
    const char* cur = ldsB[ki & 1];
    if (ki + 1 < KTILES)
      stage_b(wT2, ki + 1, ldsB[(ki + 1) & 1], tid);

    const int dr = ki / 3 - 1;
    const int ds = ki - (ki / 3) * 3 - 1;

    if (ki + 1 < KTILES) {
      const int dr2 = (ki + 1) / 3 - 1;
      const int ds2 = (ki + 1) - ((ki + 1) / 3) * 3 - 1;
      __builtin_prefetch(pc0 + (dr2 * WDIM + ds2) * 64, 0, 0);
      __builtin_prefetch(pc1 + (dr2 * WDIM + ds2) * 64, 0, 0);
    }

    v8i a0 = load_a_frag(pc0, y0, x0, dr, ds, half);
    v8i a1 = load_a_frag(pc1, y1, x1, dr, ds, half);

    v8i b[4];
#pragma unroll
    for (int nf = 0; nf < 4; ++nf)
      b[nf] = load_b_frag(cur, nWave * 64 + nf * 16 + row, half);

#pragma unroll
    for (int nf = 0; nf < 4; ++nf) {
      acc[0][nf] = __builtin_amdgcn_wmma_i32_16x16x64_iu8(
          true, a0, true, b[nf], acc[0][nf], false, false);
      acc[1][nf] = __builtin_amdgcn_wmma_i32_16x16x64_iu8(
          true, a1, true, b[nf], acc[1][nf], false, false);
    }

    if (ki + 1 < KTILES) {
      stage_wait_async();
      __syncthreads();
    }
  }

  float ax = fmaxf(alpha_x[0], 0.0f);
  float aw = fmaxf(alpha_w[0], 0.0f);
  float scale = (2.0f * ax / 254.0f) * (2.0f * aw / 254.0f);

#pragma unroll
  for (int mf = 0; mf < 2; ++mf) {
    int m_start = block_m0 + mWave * 32 + mf * 16 + half * 8;
    int n  = m_start / HW;
    int yx = m_start - n * HW;
#pragma unroll
    for (int nf = 0; nf < 4; ++nf) {
      int o = nWave * 64 + nf * 16 + row;
      float* p = out + ((long)n * COUT + o) * HW + yx;
      float4 lo = make_float4(acc[mf][nf][0] * scale, acc[mf][nf][1] * scale,
                              acc[mf][nf][2] * scale, acc[mf][nf][3] * scale);
      float4 hi = make_float4(acc[mf][nf][4] * scale, acc[mf][nf][5] * scale,
                              acc[mf][nf][6] * scale, acc[mf][nf][7] * scale);
      *(float4*)(p)     = lo;
      *(float4*)(p + 4) = hi;
    }
  }
}

extern "C" void kernel_launch(void* const* d_in, const int* in_sizes, int n_in,
                              void* d_out, int out_size, void* d_ws, size_t ws_size,
                              hipStream_t stream) {
  const float* x       = (const float*)d_in[0];
  const float* w       = (const float*)d_in[1];
  const float* alpha_x = (const float*)d_in[2];
  const float* alpha_w = (const float*)d_in[3];
  float* out = (float*)d_out;

  signed char* ixT = (signed char*)d_ws;
  signed char* wT2 = ixT + (size_t)NIMG * HW * CIN;

  quant_x_kernel<<<(NIMG * HW * CIN) / 256, 256, 0, stream>>>(x, alpha_x, ixT);
  quant_w_kernel<<<(KTILES * COUT * CIN) / 256, 256, 0, stream>>>(w, alpha_w, wT2);
  conv_wmma_kernel<<<MTOT / 128, 256, 0, stream>>>(ixT, wT2, alpha_x, alpha_w, out);
}
